// svsLanguage_deterministic_46978352283805
// MI455X (gfx1250) — matrix operand order repaired and checked by reading, not compiled, not run on hardware
//
#include <hip/hip_runtime.h>
#include <hip/hip_bf16.h>

// ---------------------------------------------------------------------------
// CDNA5 (gfx1250) WMMA implementation of the LSTM-VAE reference.
// All GEMMs (dense + recurrent LSTM gates) run on v_wmma_f32_16x16x32_f16.
// Weights are pre-packed into WMMA B-fragment-native layout (f16, padded),
// so the hot loop is: 2x b128 A-loads + 1x v16h B-load + WMMA.
// ---------------------------------------------------------------------------

typedef __attribute__((ext_vector_type(16))) _Float16 v16h;
typedef __attribute__((ext_vector_type(8)))  _Float16 v8h;
typedef __attribute__((ext_vector_type(8)))  float    v8f;

#define SEQ_LEN 100
#define BATCH   2048

__device__ __forceinline__ float sigf(float x) { return 1.0f / (1.0f + __expf(-x)); }

// A-fragment (16x32 f16, M x K): lane l holds row M=(l&15); halves 0..7 are
// K = 8*(l>>4)+0..7, halves 8..15 are K = 8*(l>>4)+16..23  -> two b128 loads.
__device__ __forceinline__ v16h load_a_frag(const _Float16* p) {
  v8h a0 = *(const v8h*)(p);
  v8h a1 = *(const v8h*)(p + 16);
  return __builtin_shufflevector(a0, a1, 0, 1, 2, 3, 4, 5, 6, 7,
                                 8, 9, 10, 11, 12, 13, 14, 15);
}

__device__ __forceinline__ v8f wmma16(v16h a, v16h b, v8f c) {
  return __builtin_amdgcn_wmma_f32_16x16x32_f16(false, a, false, b,
                                                (short)0, c, false, false);
}

// ---------------------------------------------------------------------------
// Weight packer: src is fp32 [gates*H, K] row-major (PyTorch W, used as X@W^T).
// dst is f16 fragment-packed: for B-tile (ntile over gates*Hp/16, kt over Kp/32)
// lane l, half j  ->  K = kt*32 + 8*(l>>4) + (j&7) + 16*(j>>3),  N = ntile*16 + (l&15).
// Padding (k>=K or h>=H) written as zero.
// ---------------------------------------------------------------------------
__global__ void pack_w_kernel(const float* __restrict__ src, _Float16* __restrict__ dst,
                              int K, int Kp, int H, int Hp, int gates) {
  int idx = blockIdx.x * 256 + threadIdx.x;
  int total = gates * Hp * Kp;
  if (idx >= total) return;
  int j = idx & 15;
  int l = (idx >> 4) & 31;
  int t = idx >> 9;
  int nKt = Kp >> 5;
  int kt = t % nKt;
  int ntile = t / nKt;
  int k = (kt << 5) + ((l >> 4) << 3) + (j & 7) + ((j >> 3) << 4);
  int n = (ntile << 4) + (l & 15);
  int gate = n / Hp;
  int h = n - gate * Hp;
  float v = 0.0f;
  if (k < K && h < H) v = src[(size_t)(gate * H + h) * K + k];
  dst[idx] = (_Float16)v;
}

__global__ void pack_b_kernel(const float* __restrict__ b1, const float* __restrict__ b2,
                              float* __restrict__ dst, int H, int Hp, int gates) {
  int idx = blockIdx.x * 256 + threadIdx.x;
  if (idx >= gates * Hp) return;
  int gate = idx / Hp;
  int h = idx - gate * Hp;
  float v = 0.0f;
  if (h < H) {
    v = b1[gate * H + h];
    if (b2) v += b2[gate * H + h];
  }
  dst[idx] = v;
}

__global__ void f32_to_f16_kernel(const float* __restrict__ src,
                                  _Float16* __restrict__ dst, int n) {
  int i = blockIdx.x * 256 + threadIdx.x;
  if (i < n) dst[i] = (_Float16)src[i];
}

// ---------------------------------------------------------------------------
// Fused LSTM cell: gates = X@Wih^T + Hin@Whh^T + bias, then
// c' = sig(f)*c + sig(i)*tanh(g); h' = sig(o)*tanh(c').
// Each wave: 16x16 (batch x hidden) tile, 4 accumulators (i,f,g,o).
// grid = (Hp/16, B/128), block = 256 (8 wave32s).
// ---------------------------------------------------------------------------
__global__ __launch_bounds__(256) void lstm_cell_wmma(
    const _Float16* __restrict__ X, int ldx, int Kx,
    const _Float16* __restrict__ Hin, int Kh,
    const _Float16* __restrict__ Wih,
    const _Float16* __restrict__ Whh,
    const float* __restrict__ bias,
    float* __restrict__ C,
    _Float16* __restrict__ Hout,
    int Hp, int Hlog,
    _Float16* __restrict__ exp16, int e16ld, int e16cs, int e16off, int do_relu,
    float* __restrict__ exp32, int e32ld) {
  const int lane = threadIdx.x & 31;
  const int wv = threadIdx.x >> 5;
  const int nt = blockIdx.x;
  const int mt = blockIdx.y * 8 + wv;
  const int row0 = mt << 4;
  const int grp = lane >> 4;
  const int arow = lane & 15;
  const int col = lane & 15;
  const int hT = Hp >> 4;
  const int nKx = Kx >> 5;
  const int nKh = Kh >> 5;

  v8f acc[4] = {};

  const _Float16* arow_ptr = X + (size_t)(row0 + arow) * ldx + (grp << 3);
  for (int kt = 0; kt < nKx; ++kt) {
    v16h a = load_a_frag(arow_ptr + (kt << 5));
#pragma unroll
    for (int g = 0; g < 4; ++g) {
      const _Float16* bp = Wih + ((((size_t)(g * hT + nt)) * nKx + kt) << 9) + (lane << 4);
      __builtin_prefetch(bp + 512, 0, 1);  // next K-tile -> global_prefetch_b8
      v16h b = *(const v16h*)bp;
      acc[g] = wmma16(a, b, acc[g]);
    }
  }
  const _Float16* hrow_ptr = Hin + (size_t)(row0 + arow) * Kh + (grp << 3);
  for (int kt = 0; kt < nKh; ++kt) {
    v16h a = load_a_frag(hrow_ptr + (kt << 5));
#pragma unroll
    for (int g = 0; g < 4; ++g) {
      const _Float16* bp = Whh + ((((size_t)(g * hT + nt)) * nKh + kt) << 9) + (lane << 4);
      __builtin_prefetch(bp + 512, 0, 1);
      v16h b = *(const v16h*)bp;
      acc[g] = wmma16(a, b, acc[g]);
    }
  }

  const int colg = (nt << 4) + col;
  const float bI = bias[colg];
  const float bF = bias[Hp + colg];
  const float bG = bias[2 * Hp + colg];
  const float bO = bias[3 * Hp + colg];

#pragma unroll
  for (int r = 0; r < 8; ++r) {
    const int row = row0 + r + (grp << 3);  // C/D layout: lanes16-31 -> M=r+8
    const size_t idx = (size_t)row * Hp + colg;
    float iv = sigf(acc[0][r] + bI);
    float fv = sigf(acc[1][r] + bF);
    float gv = tanhf(acc[2][r] + bG);
    float ov = sigf(acc[3][r] + bO);
    float cn = fv * C[idx] + iv * gv;
    float hn = ov * tanhf(cn);
    C[idx] = cn;
    Hout[idx] = (_Float16)hn;
    if (colg < Hlog) {
      if (exp16) {
        float e = do_relu ? fmaxf(hn, 0.0f) : hn;
        exp16[(size_t)row * e16ld + (size_t)colg * e16cs + e16off] = (_Float16)e;
      }
      if (exp32) exp32[(size_t)row * e32ld + colg] = hn;
    }
  }
}

// ---------------------------------------------------------------------------
// Dense GEMM: Y = act(X @ W^T + b); W fragment-packed (gates=1).
// ---------------------------------------------------------------------------
__global__ __launch_bounds__(256) void gemm_wmma(
    const _Float16* __restrict__ X, int ldx, int K,
    const _Float16* __restrict__ W, const float* __restrict__ bias,
    _Float16* __restrict__ Y16, int Np, int Nlog, int do_relu,
    float* __restrict__ Y32, int y32ld) {
  const int lane = threadIdx.x & 31;
  const int wv = threadIdx.x >> 5;
  const int nt = blockIdx.x;
  const int mt = blockIdx.y * 8 + wv;
  const int row0 = mt << 4;
  const int grp = lane >> 4;
  const int arow = lane & 15;
  const int col = lane & 15;
  const int nKt = K >> 5;

  v8f acc = {};
  const _Float16* arow_ptr = X + (size_t)(row0 + arow) * ldx + (grp << 3);
  for (int kt = 0; kt < nKt; ++kt) {
    v16h a = load_a_frag(arow_ptr + (kt << 5));
    const _Float16* bp = W + (((size_t)nt * nKt + kt) << 9) + (lane << 4);
    __builtin_prefetch(bp + 512, 0, 1);
    v16h b = *(const v16h*)bp;
    acc = wmma16(a, b, acc);
  }
  const int colg = (nt << 4) + col;
  const float bb = bias[colg];
#pragma unroll
  for (int r = 0; r < 8; ++r) {
    const int row = row0 + r + (grp << 3);
    float y = acc[r] + bb;
    if (do_relu) y = fmaxf(y, 0.0f);
    if (Y16) Y16[(size_t)row * Np + colg] = (_Float16)y;
    if (Y32 && colg < Nlog) Y32[(size_t)row * y32ld + colg] = y;
  }
}

// ---------------------------------------------------------------------------
// Host orchestration.
// ---------------------------------------------------------------------------
extern "C" void kernel_launch(void* const* d_in, const int* in_sizes, int n_in,
                              void* d_out, int out_size, void* d_ws, size_t ws_size,
                              hipStream_t stream) {
  (void)in_sizes; (void)n_in; (void)out_size;

  const float* x        = (const float*)d_in[0];
  const float* fc1_wih  = (const float*)d_in[1];
  const float* fc1_whh  = (const float*)d_in[2];
  const float* fc1_bih  = (const float*)d_in[3];
  const float* fc1_bhh  = (const float*)d_in[4];
  const float* fc21_wih = (const float*)d_in[5];
  const float* fc21_whh = (const float*)d_in[6];
  const float* fc21_bih = (const float*)d_in[7];
  const float* fc21_bhh = (const float*)d_in[8];
  const float* lin1_w   = (const float*)d_in[9];
  const float* lin1_b   = (const float*)d_in[10];
  const float* lin2_w   = (const float*)d_in[11];
  const float* lin2_b   = (const float*)d_in[12];
  const float* idlin_w  = (const float*)d_in[13];
  const float* idlin_b  = (const float*)d_in[14];
  const float* down1_w  = (const float*)d_in[15];
  const float* down1_b  = (const float*)d_in[16];
  const float* down2_w  = (const float*)d_in[17];
  const float* down2_b  = (const float*)d_in[18];
  const float* l3_wih   = (const float*)d_in[19];
  const float* l3_whh   = (const float*)d_in[20];
  const float* l3_bih   = (const float*)d_in[21];
  const float* l3_bhh   = (const float*)d_in[22];
  const float* l4_wih   = (const float*)d_in[23];
  const float* l4_whh   = (const float*)d_in[24];
  const float* l4_bih   = (const float*)d_in[25];
  const float* l4_bhh   = (const float*)d_in[26];
  const float* l5_wih   = (const float*)d_in[27];
  const float* l5_whh   = (const float*)d_in[28];
  const float* l5_bih   = (const float*)d_in[29];
  const float* l5_bhh   = (const float*)d_in[30];

  char* base = (char*)d_ws;
  size_t cur = 0;
  auto alloc  = [&](size_t bytes) -> char* {
    char* p = base + cur;
    cur = (cur + bytes + 255) & ~(size_t)255;
    return p;
  };
  auto allocH = [&](size_t n) -> _Float16* { return (_Float16*)alloc(n * sizeof(_Float16)); };
  auto allocF = [&](size_t n) -> float*    { return (float*)alloc(n * sizeof(float)); };

  // Padded dims: MID_I 60->64, LATENT 20->32, V_LAT 50->64, flatK 2000->2016.
  // Packed weights (f16, fragment layout).
  _Float16* wfc1ih  = allocH((size_t)4 * 64 * 64);
  _Float16* wfc1hh  = allocH((size_t)4 * 64 * 64);
  _Float16* wfc21ih = allocH((size_t)4 * 32 * 64);
  _Float16* wfc21hh = allocH((size_t)4 * 32 * 32);
  _Float16* wlin1   = allocH((size_t)800 * 2016);
  _Float16* wlin2   = allocH((size_t)64 * 800);
  _Float16* widlin  = allocH((size_t)64 * 64);
  _Float16* wdown1  = allocH((size_t)32 * 64);
  _Float16* wdown2  = allocH((size_t)32 * 512);
  _Float16* wl3ih   = allocH((size_t)4 * 64 * 32);
  _Float16* wl3hh   = allocH((size_t)4 * 64 * 64);
  _Float16* wl4ih   = allocH((size_t)4 * 800 * 64);
  _Float16* wl4hh   = allocH((size_t)4 * 800 * 800);
  _Float16* wl5ih   = allocH((size_t)4 * 512 * 800);
  _Float16* wl5hh   = allocH((size_t)4 * 512 * 512);
  // Combined padded biases (f32).
  float* bfc1   = allocF(4 * 64);
  float* bfc21  = allocF(4 * 32);
  float* blin1  = allocF(800);
  float* blin2  = allocF(64);
  float* bidlin = allocF(64);
  float* bdown1 = allocF(32);
  float* bdown2 = allocF(32);
  float* bl3    = allocF(4 * 64);
  float* bl4    = allocF(4 * 800);
  float* bl5    = allocF(4 * 512);
  // Activations / states.
  _Float16* xf16    = allocH((size_t)SEQ_LEN * BATCH * 64);
  _Float16* h1buf   = allocH((size_t)BATCH * 64);
  _Float16* hfc1a   = allocH((size_t)BATCH * 64);
  _Float16* hfc1b   = allocH((size_t)BATCH * 64);
  float*    cfc1    = allocF((size_t)BATCH * 64);
  _Float16* hfc21a  = allocH((size_t)BATCH * 32);
  _Float16* hfc21b  = allocH((size_t)BATCH * 32);
  float*    cfc21   = allocF((size_t)BATCH * 32);
  _Float16* flatb   = allocH((size_t)BATCH * 2016);
  _Float16* mu1     = allocH((size_t)BATCH * 800);
  _Float16* mu16    = allocH((size_t)BATCH * 64);  // also decoder h_t ping buffer
  _Float16* htB     = allocH((size_t)BATCH * 64);
  float*    c_t     = allocF((size_t)BATCH * 64);
  _Float16* h2a     = allocH((size_t)BATCH * 800);
  _Float16* h2b     = allocH((size_t)BATCH * 800);
  float*    c2      = allocF((size_t)BATCH * 800);
  _Float16* h3a     = allocH((size_t)BATCH * 512);
  _Float16* h3b     = allocH((size_t)BATCH * 512);
  float*    c3      = allocF((size_t)BATCH * 512);
  _Float16* outb    = allocH((size_t)BATCH * 32);

  // Zero states + padding lanes once (memset node is graph-capturable).
  size_t zbytes = cur < ws_size ? cur : ws_size;
  hipMemsetAsync(d_ws, 0, zbytes, stream);

  auto packW = [&](const float* src, _Float16* dst, int K, int Kp, int H, int Hp, int gates) {
    int total = gates * Hp * Kp;
    pack_w_kernel<<<dim3((total + 255) / 256), dim3(256), 0, stream>>>(src, dst, K, Kp, H, Hp, gates);
  };
  auto packB = [&](const float* b1, const float* b2, float* dst, int H, int Hp, int gates) {
    int total = gates * Hp;
    pack_b_kernel<<<dim3((total + 255) / 256), dim3(256), 0, stream>>>(b1, b2, dst, H, Hp, gates);
  };
  auto lstm = [&](const _Float16* X, int ldx, int Kx, const _Float16* Hin, int Kh,
                  const _Float16* Wih, const _Float16* Whh, const float* bias,
                  float* C, _Float16* Hout, int Hp, int Hlog,
                  _Float16* e16, int e16ld, int e16cs, int e16off, int relu,
                  float* e32, int e32ld) {
    lstm_cell_wmma<<<dim3(Hp >> 4, BATCH >> 7), dim3(256), 0, stream>>>(
        X, ldx, Kx, Hin, Kh, Wih, Whh, bias, C, Hout, Hp, Hlog,
        e16, e16ld, e16cs, e16off, relu, e32, e32ld);
  };
  auto gemm = [&](const _Float16* X, int ldx, int K, const _Float16* W, const float* bias,
                  _Float16* Y16, int Np, int Nlog, int relu, float* Y32, int y32ld) {
    gemm_wmma<<<dim3(Np >> 4, BATCH >> 7), dim3(256), 0, stream>>>(
        X, ldx, K, W, bias, Y16, Np, Nlog, relu, Y32, y32ld);
  };

  // ---- pack all weights / biases, convert x to f16 ----
  packW(fc1_wih,  wfc1ih,  64,  64,  60, 64, 4);
  packW(fc1_whh,  wfc1hh,  60,  64,  60, 64, 4);
  packW(fc21_wih, wfc21ih, 60,  64,  20, 32, 4);
  packW(fc21_whh, wfc21hh, 20,  32,  20, 32, 4);
  packW(lin1_w,   wlin1,   2000, 2016, 800, 800, 1);
  packW(lin2_w,   wlin2,   800, 800,  50, 64, 1);
  packW(idlin_w,  widlin,  50,  64,   50, 64, 1);
  packW(down1_w,  wdown1,  50,  64,   20, 32, 1);
  packW(down2_w,  wdown2,  512, 512,  20, 32, 1);
  packW(l3_wih,   wl3ih,   20,  32,   50, 64, 4);
  packW(l3_whh,   wl3hh,   50,  64,   50, 64, 4);
  packW(l4_wih,   wl4ih,   50,  64,   800, 800, 4);
  packW(l4_whh,   wl4hh,   800, 800,  800, 800, 4);
  packW(l5_wih,   wl5ih,   800, 800,  512, 512, 4);
  packW(l5_whh,   wl5hh,   512, 512,  512, 512, 4);

  packB(fc1_bih, fc1_bhh, bfc1, 60, 64, 4);
  packB(fc21_bih, fc21_bhh, bfc21, 20, 32, 4);
  packB(lin1_b, nullptr, blin1, 800, 800, 1);
  packB(lin2_b, nullptr, blin2, 50, 64, 1);
  packB(idlin_b, nullptr, bidlin, 50, 64, 1);
  packB(down1_b, nullptr, bdown1, 20, 32, 1);
  packB(down2_b, nullptr, bdown2, 20, 32, 1);
  packB(l3_bih, l3_bhh, bl3, 50, 64, 4);
  packB(l4_bih, l4_bhh, bl4, 800, 800, 4);
  packB(l5_bih, l5_bhh, bl5, 512, 512, 4);

  {
    int n = SEQ_LEN * BATCH * 64;
    f32_to_f16_kernel<<<dim3((n + 255) / 256), dim3(256), 0, stream>>>(x, xf16, n);
  }

  // ---- encoder: fc1 (relu-exported) -> fc21 (scattered into flatten buf) ----
  _Float16* hf1[2] = {hfc1a, hfc1b}; int p1 = 0;
  _Float16* hf2[2] = {hfc21a, hfc21b}; int pe = 0;
  for (int s = 0; s < SEQ_LEN; ++s) {
    lstm(xf16 + (size_t)s * BATCH * 64, 64, 64, hf1[p1], 64,
         wfc1ih, wfc1hh, bfc1, cfc1, hf1[1 - p1], 64, 60,
         h1buf, 64, 1, 0, /*relu=*/1, nullptr, 0);
    p1 ^= 1;
    lstm(h1buf, 64, 64, hf2[pe], 32,
         wfc21ih, wfc21hh, bfc21, cfc21, hf2[1 - pe], 32, 20,
         flatb, 2016, SEQ_LEN, s, /*relu=*/0, nullptr, 0);
    pe ^= 1;
  }

  // ---- latent head ----
  float* mu_out = (float*)d_out + (size_t)SEQ_LEN * BATCH * 512;
  gemm(flatb, 2016, 2016, wlin1, blin1, mu1, 800, 800, 1, nullptr, 0);
  gemm(mu1, 800, 800, wlin2, blin2, mu16, 64, 50, 1, mu_out, 50);
  gemm(mu16, 64, 64, widlin, bidlin, nullptr, 64, 64, 0, c_t, 64);   // c_t init
  gemm(mu16, 64, 64, wdown1, bdown1, outb, 32, 20, 0, nullptr, 0);   // out init

  // ---- decoder: l3 -> l4 -> l5 -> down2, 100 steps ----
  _Float16* ht[2] = {mu16, htB}; int pt = 0;   // h_t starts at mu
  _Float16* h2[2] = {h2a, h2b};  int p2 = 0;
  _Float16* h3[2] = {h3a, h3b};  int p3 = 0;
  for (int s = 0; s < SEQ_LEN; ++s) {
    lstm(outb, 32, 32, ht[pt], 64, wl3ih, wl3hh, bl3, c_t, ht[1 - pt], 64, 50,
         nullptr, 0, 0, 0, 0, nullptr, 0);
    pt ^= 1;
    lstm(ht[pt], 64, 64, h2[p2], 800, wl4ih, wl4hh, bl4, c2, h2[1 - p2], 800, 800,
         nullptr, 0, 0, 0, 0, nullptr, 0);
    p2 ^= 1;
    lstm(h2[p2], 800, 800, h3[p3], 512, wl5ih, wl5hh, bl5, c3, h3[1 - p3], 512, 512,
         nullptr, 0, 0, 0, 0, (float*)d_out + (size_t)s * BATCH * 512, 512);
    p3 ^= 1;
    gemm(h3[p3], 512, 512, wdown2, bdown2, outb, 32, 20, 0, nullptr, 0);
  }
}
